// CstPcd_36369783062855
// MI455X (gfx1250) — matrix operand order repaired and checked by reading, not compiled, not run on hardware
//
#include <hip/hip_runtime.h>
#include <hip/hip_bf16.h>
#include <float.h>
#include <math.h>

// ---------------------------------------------------------------------------
// CDNA5 / gfx1250 implementation of the PointNet++-style pipeline.
//  - MLP GEMMs:  v_wmma_f32_16x16x32_bf16, B staged via TENSOR_LOAD_TO_LDS
//                (TDM) into LDS once per workgroup, A as global b128 loads
//  - KNN dists:  v_wmma_f32_16x16x4_f32  (xyz padded K=3 -> 4, exact f32)
//  - 16-row x 2048-col distance block staged in 128KB LDS (320KB/WGP on CDNA5)
//  - Deterministic BN reductions (no float atomics) for graph-replay safety
// ---------------------------------------------------------------------------

typedef __attribute__((ext_vector_type(16))) __bf16 v16bf;
typedef __attribute__((ext_vector_type(8)))  __bf16 v8bf;
typedef __attribute__((ext_vector_type(8)))  float  v8f;
typedef __attribute__((ext_vector_type(2)))  float  v2f;
typedef __attribute__((ext_vector_type(4)))  unsigned int u32x4;
typedef __attribute__((ext_vector_type(8)))  int    i32x8;
typedef __attribute__((ext_vector_type(4)))  int    i32x4;

#define BN_EPS 1e-5f

__device__ __forceinline__ float lrelu(float x) { return x > 0.f ? x : 0.2f * x; }

// ---------------------------------------------------------------------------
// KNN: one block = (batch b, 16-row tile). Phase 1: WMMA f32 16x16x4 distance
// tiles into LDS. Phase 2: per-row iterative wave-parallel argmin (k smallest).
// ---------------------------------------------------------------------------
__global__ void knn_kernel(const float* __restrict__ xyz, int* __restrict__ idx_out,
                           int n, int ncols32, int k)
{
    extern __shared__ float sdist[];            // 16 * ncols32 floats
    const int b    = blockIdx.x;
    const int rt   = blockIdx.y;
    const int tid  = threadIdx.x;
    const int lane = tid & 31;
    const int wave = tid >> 5;                  // 0..7
    const int hi   = lane >> 4;
    const int lm   = lane & 15;
    const int rowbase = rt * 16;
    const float* P = xyz + (size_t)b * n * 3;

    // A fragment: rows of this tile, coords (x,y | z,0) split across lane halves
    int arow = rowbase + lm; if (arow >= n) arow = n - 1;
    float a0 = P[arow * 3 + (hi ? 2 : 0)];
    float a1 = hi ? 0.f : P[arow * 3 + 1];
    v2f av; av.x = a0; av.y = a1;
    float sqa_p = a0 * a0 + a1 * a1;
    float sqa = sqa_p + __shfl_xor(sqa_p, 16, 32);   // |x_m|^2 for m = lane&15

    // hoisted: |x|^2 for the 8 C-fragment rows this lane will see
    float sqa_r[8];
#pragma unroll
    for (int r = 0; r < 8; ++r) sqa_r[r] = __shfl(sqa, r + 8 * hi, 32);

    const int ntile = ncols32 >> 4;
    for (int ct = wave; ct < ntile; ct += 8) {
        int col  = ct * 16 + lm;
        int ccol = col < n ? col : n - 1;
        float b0 = P[ccol * 3 + (hi ? 2 : 0)];
        float b1 = hi ? 0.f : P[ccol * 3 + 1];
        v2f bv; bv.x = b0; bv.y = b1;
        float sqb_p = b0 * b0 + b1 * b1;
        float sqb = sqb_p + __shfl_xor(sqb_p, 16, 32);
        v8f c = {};
        c = __builtin_amdgcn_wmma_f32_16x16x4_f32(false, av, false, bv,
                                                  (short)0, c, false, false);
#pragma unroll
        for (int r = 0; r < 8; ++r) {
            int rr = r + 8 * hi;
            float d = (col < n && (rowbase + rr) < n) ? (sqa_r[r] + sqb - 2.f * c[r])
                                                      : FLT_MAX;
            sdist[rr * ncols32 + col] = d;
        }
    }
    __syncthreads();

    // selection: wave w handles rows w and w+8 of the tile
    const int nt = ncols32 >> 5;                 // <= 64
    for (int rsel = wave; rsel < 16; rsel += 8) {
        int row = rowbase + rsel;
        if (row >= n) continue;
        float v[64];
        for (int t = 0; t < nt; ++t) v[t] = sdist[rsel * ncols32 + lane + 32 * t];
        for (int it = 0; it < k; ++it) {
            float bestv = FLT_MAX; int besti = n;
            for (int t = 0; t < nt; ++t) {
                int ci = lane + 32 * t;
                if (v[t] < bestv || (v[t] == bestv && ci < besti)) { bestv = v[t]; besti = ci; }
            }
#pragma unroll
            for (int off = 16; off >= 1; off >>= 1) {
                float ov = __shfl_xor(bestv, off, 32);
                int   oi = __shfl_xor(besti, off, 32);
                if (ov < bestv || (ov == bestv && oi < besti)) { bestv = ov; besti = oi; }
            }
            if (lane == 0) idx_out[((size_t)b * n + row) * k + it] = besti;
            if ((besti & 31) == lane) v[besti >> 5] = FLT_MAX;
        }
    }
}

// ---------------------------------------------------------------------------
// Farthest point sampling: one block per batch, sequential nc iterations.
// ---------------------------------------------------------------------------
__global__ void fps_kernel(const float* __restrict__ xyz, int* __restrict__ out,
                           int n, int nc)
{
    const int b = blockIdx.x, tid = threadIdx.x;
    __shared__ float sval[256];
    __shared__ int   sidx[256];
    __shared__ int   sfar;
    const float* P = xyz + (size_t)b * n * 3;
    float dmin[8];
#pragma unroll
    for (int t = 0; t < 8; ++t) dmin[t] = 1e10f;
    if (tid == 0) sfar = 0;
    __syncthreads();
    for (int i = 0; i < nc; ++i) {
        int far = sfar;
        if (tid == 0) out[(size_t)b * nc + i] = far;
        float cx = P[far * 3], cy = P[far * 3 + 1], cz = P[far * 3 + 2];
        float bv = -1.f; int bi = 0;
        for (int t = 0; t < 8; ++t) {
            int p = tid + 256 * t;
            if (p < n) {
                float dx = P[p * 3] - cx, dy = P[p * 3 + 1] - cy, dz = P[p * 3 + 2] - cz;
                float d = dx * dx + dy * dy + dz * dz;
                if (d < dmin[t]) dmin[t] = d;
                if (dmin[t] > bv) { bv = dmin[t]; bi = p; }
            }
        }
        sval[tid] = bv; sidx[tid] = bi;
        __syncthreads();
        for (int s = 128; s > 0; s >>= 1) {
            if (tid < s) {
                float ov = sval[tid + s]; int oi = sidx[tid + s];
                if (ov > sval[tid] || (ov == sval[tid] && oi < sidx[tid])) {
                    sval[tid] = ov; sidx[tid] = oi;
                }
            }
            __syncthreads();
        }
        if (tid == 0) sfar = sidx[0];
        __syncthreads();
    }
}

// ---------------------------------------------------------------------------
// Build grouped WMMA A-matrix rows: [group_fea | xyz_rel | 0-pad] as bf16.
// ---------------------------------------------------------------------------
__global__ void gather_group_kernel(const float* __restrict__ xyz, const float* __restrict__ fea,
                                    const int* __restrict__ knn_idx, const int* __restrict__ fps_idx,
                                    __bf16* __restrict__ A, float* __restrict__ cxyz,
                                    int n, int nc, int k, int f, int Kpad, long long Mpad, int bs)
{
    long long r = (long long)blockIdx.x * blockDim.x + threadIdx.x;
    if (r >= Mpad) return;
    long long M = (long long)bs * nc * k;
    __bf16* arow = A + r * Kpad;
    if (r >= M) { for (int c = 0; c < Kpad; ++c) arow[c] = (__bf16)0.f; return; }
    int b   = (int)(r / ((long long)nc * k));
    int rem = (int)(r % ((long long)nc * k));
    int cI = rem / k, j = rem % k;
    int center = fps_idx[(size_t)b * nc + cI];
    int gi     = knn_idx[((size_t)b * n + center) * k + j];
    const float* Fb = fea + ((size_t)b * n + gi) * f;
    const float* Pg = xyz + ((size_t)b * n + gi) * 3;
    const float* Pc = xyz + ((size_t)b * n + center) * 3;
    int c = 0;
    for (; c < f; ++c) arow[c] = (__bf16)Fb[c];
    for (int d = 0; d < 3; ++d, ++c) arow[c] = (__bf16)(Pg[d] - Pc[d]);
    for (; c < Kpad; ++c) arow[c] = (__bf16)0.f;
    if (j == 0) {
        float* o = cxyz + ((size_t)b * nc + cI) * 3;
        o[0] = Pc[0]; o[1] = Pc[1]; o[2] = Pc[2];
    }
}

// ---------------------------------------------------------------------------
// WMMA GEMM, 8 waves/block, one 16x16 tile per wave, shared N-block.
// B is pre-packed fragment-major: Bp[(kt*Npad + n)*32 + h*16 + j]
//   = W[kt*32 + 16h + j][n], so one lane's fragment half is 16 contiguous
//   bf16. The whole (colblock x Kpad) slab (<=12KB) is staged into LDS by the
//   Tensor Data Mover once per workgroup, then read back with ds_load_b128.
// A fragment = two adjacent 16B runs per lane (contiguous k = 16h + s, matching the B pack).
// ---------------------------------------------------------------------------
__global__ void wmma_gemm_kernel(const __bf16* __restrict__ A, const __bf16* __restrict__ Bp,
                                 const float* __restrict__ bias, float* __restrict__ Y,
                                 int Kpad, int Npad, int Cout, int Mtiles)
{
    __shared__ __bf16 sB[6144];                 // (Kpad/32) * 512 bf16, Kpad<=384
    const int tid  = threadIdx.x;
    const int lane = tid & 31;
    const int wave = tid >> 5;                  // 0..7
    const int hi = lane >> 4, lm = lane & 15;
    const int colbase = blockIdx.y * 16;
    const int nkt = Kpad >> 5;

    if (wave == 0) {
        // --- Tensor DMA: 2D tile, rows = k-tiles, 512 elems (1KB) per row ---
        unsigned long long ga = (unsigned long long)(size_t)Bp
                              + (unsigned long long)colbase * 64ull;   // colbase*32 elems * 2B
        unsigned dim0 = (unsigned)Npad * 32u;       // tensor row length (elems)
        u32x4 g0;
        g0[0] = 1u;                                  // count=1 (valid descriptor)
        g0[1] = 0u;                                  // lds_addr: sB is the only LDS var -> offset 0
        g0[2] = (unsigned)ga;                        // global_addr[31:0]
        g0[3] = (unsigned)((ga >> 32) & 0x01FFFFFFu) | 0x80000000u;  // addr[56:32] | type=2
        i32x8 g1;
        g1[0] = (int)(1u << 16);                     // data_size=1 (2 bytes)
        g1[1] = (int)((dim0 & 0xFFFFu) << 16);       // tensor_dim0[15:0] @bits48
        g1[2] = (int)((dim0 >> 16) | (((unsigned)nkt & 0xFFFFu) << 16)); // dim0 hi | dim1 lo
        g1[3] = (int)(512u << 16);                   // dim1 hi=0 | tile_dim0=512
        g1[4] = (int)(unsigned)nkt;                  // tile_dim1=nkt, tile_dim2=0
        g1[5] = (int)dim0;                           // tensor_dim0_stride[31:0]
        g1[6] = 0;                                   // stride0 hi | stride1 lo
        g1[7] = 0;                                   // stride1 hi
        i32x4 z4 = {};
#if defined(__clang_major__) && (__clang_major__ >= 23)
        i32x8 z8 = {};
        __builtin_amdgcn_tensor_load_to_lds(g0, g1, z4, z4, z8, 0);
#else
        __builtin_amdgcn_tensor_load_to_lds(g0, g1, z4, z4, 0);
#endif
        __builtin_amdgcn_s_wait_tensorcnt(0);
    }
    __syncthreads();

    const int mt = blockIdx.x * 8 + wave;
    if (mt >= Mtiles) return;
    const size_t rowbase = (size_t)mt * 16;
    const __bf16* arow = A + (rowbase + lm) * (size_t)Kpad;
    const int sfragbase = lm * 32 + hi * 16;

    v8f acc = {};
    for (int kt = 0; kt < nkt; ++kt) {
        const int k0 = kt << 5;
        if (kt + 1 < nkt) __builtin_prefetch(arow + k0 + 32, 0, 0);
        v8bf alo = *reinterpret_cast<const v8bf*>(arow + k0 + 16 * hi);
        v8bf ahi = *reinterpret_cast<const v8bf*>(arow + k0 + 16 * hi + 8);
        const v8bf* sp = reinterpret_cast<const v8bf*>(sB + kt * 512 + sfragbase);
        v8bf blo = sp[0];
        v8bf bhi = sp[1];
        v16bf af, bf;
#pragma unroll
        for (int j = 0; j < 8; ++j) {
            af[j] = alo[j]; af[8 + j] = ahi[j];
            bf[j] = blo[j]; bf[8 + j] = bhi[j];
        }
        acc = __builtin_amdgcn_wmma_f32_16x16x32_bf16(false, af, false, bf,
                                                      (short)0, acc, false, false);
    }
    const int col = colbase + lm;
    const float bb = (col < Cout) ? bias[col] : 0.f;
#pragma unroll
    for (int r = 0; r < 8; ++r)
        Y[(rowbase + r + 8 * hi) * (size_t)Npad + col] = acc[r] + bb;
}

// ---------------------------------------------------------------------------
// Deterministic per-channel mean/var (one workgroup per channel).
// ---------------------------------------------------------------------------
__global__ void bn_stats_kernel(const float* __restrict__ Y, float* __restrict__ mean,
                                float* __restrict__ var, long long M, int Npad)
{
    const int ch = blockIdx.x, tid = threadIdx.x;
    __shared__ float s1[256], s2[256];
    float a = 0.f, b = 0.f;
    for (long long r = tid; r < M; r += 256) {
        float x = Y[(size_t)r * Npad + ch]; a += x; b += x * x;
    }
    s1[tid] = a; s2[tid] = b; __syncthreads();
    for (int s = 128; s > 0; s >>= 1) {
        if (tid < s) { s1[tid] += s1[tid + s]; s2[tid] += s2[tid + s]; }
        __syncthreads();
    }
    if (tid == 0) {
        float m = s1[0] / (float)M;
        mean[ch] = m;
        var[ch]  = s2[0] / (float)M - m * m;
    }
}

__global__ void bn_act_bf16_kernel(const float* __restrict__ Y, const float* __restrict__ mean,
                                   const float* __restrict__ var, const float* __restrict__ g,
                                   const float* __restrict__ be, __bf16* __restrict__ A,
                                   long long M, long long Mpad, int Npad, int Cout, int KpadDst)
{
    long long r = (long long)blockIdx.x * blockDim.x + threadIdx.x;
    if (r >= Mpad) return;
    __bf16* arow = A + r * (size_t)KpadDst;
    if (r >= M) { for (int c = 0; c < KpadDst; ++c) arow[c] = (__bf16)0.f; return; }
    const float* yrow = Y + r * (size_t)Npad;
    int c = 0;
    for (; c < Cout; ++c) {
        float x  = yrow[c];
        float yv = g[c] * (x - mean[c]) * rsqrtf(var[c] + BN_EPS) + be[c];
        arow[c] = (__bf16)lrelu(yv);
    }
    for (; c < KpadDst; ++c) arow[c] = (__bf16)0.f;
}

__global__ void bn_act_f32_kernel(float* __restrict__ Y, const float* __restrict__ mean,
                                  const float* __restrict__ var, const float* __restrict__ g,
                                  const float* __restrict__ be, long long M, int Npad, int Cout)
{
    long long idx = (long long)blockIdx.x * blockDim.x + threadIdx.x;
    if (idx >= M * (long long)Cout) return;
    long long r = idx / Cout; int c = (int)(idx % Cout);
    float x  = Y[(size_t)r * Npad + c];
    float yv = g[c] * (x - mean[c]) * rsqrtf(var[c] + BN_EPS) + be[c];
    Y[(size_t)r * Npad + c] = lrelu(yv);
}

__global__ void max_k_kernel(const float* __restrict__ Y, float* __restrict__ fea,
                             long long bsnc, int k, int Npad, int Cout)
{
    long long idx = (long long)blockIdx.x * blockDim.x + threadIdx.x;
    if (idx >= bsnc * (long long)Cout) return;
    long long rc = idx / Cout; int c = (int)(idx % Cout);
    float m = -FLT_MAX;
    for (int j = 0; j < k; ++j) {
        float x = Y[((size_t)rc * k + j) * Npad + c];
        m = x > m ? x : m;
    }
    fea[(size_t)rc * Cout + c] = m;
}

// ---------------------------------------------------------------------------
// 3-NN inverse-distance interpolation + concat, emitted directly as bf16 A.
// ---------------------------------------------------------------------------
__global__ void three_interp_kernel(const float* __restrict__ xyz1, const float* __restrict__ xyz2,
                                    const float* __restrict__ fea1, const float* __restrict__ fea2,
                                    __bf16* __restrict__ A,
                                    int n1, int n2, int f1, int f2, int Kpad,
                                    long long Mpad, int bs)
{
    long long r = (long long)blockIdx.x * blockDim.x + threadIdx.x;
    if (r >= Mpad) return;
    __bf16* arow = A + r * (size_t)Kpad;
    long long M = (long long)bs * n1;
    if (r >= M) { for (int c = 0; c < Kpad; ++c) arow[c] = (__bf16)0.f; return; }
    int b = (int)(r / n1);
    const float* p = xyz1 + r * 3;
    float px = p[0], py = p[1], pz = p[2];
    const float* Q = xyz2 + (size_t)b * n2 * 3;
    float d0 = FLT_MAX, d1 = FLT_MAX, d2 = FLT_MAX; int i0 = 0, i1 = 0, i2 = 0;
    for (int j = 0; j < n2; ++j) {
        float dx = Q[j * 3] - px, dy = Q[j * 3 + 1] - py, dz = Q[j * 3 + 2] - pz;
        float d = dx * dx + dy * dy + dz * dz;
        if      (d < d0) { d2 = d1; i2 = i1; d1 = d0; i1 = i0; d0 = d; i0 = j; }
        else if (d < d1) { d2 = d1; i2 = i1; d1 = d;  i1 = j; }
        else if (d < d2) { d2 = d;  i2 = j; }
    }
    float w0 = 1.f / (d0 + 1e-8f), w1 = 1.f / (d1 + 1e-8f), w2 = 1.f / (d2 + 1e-8f);
    float ws = w0 + w1 + w2; w0 /= ws; w1 /= ws; w2 /= ws;
    const float* F0 = fea2 + ((size_t)b * n2 + i0) * f2;
    const float* F1 = fea2 + ((size_t)b * n2 + i1) * f2;
    const float* F2 = fea2 + ((size_t)b * n2 + i2) * f2;
    const float* Fa = fea1 + r * (size_t)f1;
    int c = 0;
    for (; c < f1; ++c) arow[c] = (__bf16)Fa[c];
    for (int t = 0; t < f2; ++t, ++c)
        arow[c] = (__bf16)(w0 * F0[t] + w1 * F1[t] + w2 * F2[t]);
    for (; c < Kpad; ++c) arow[c] = (__bf16)0.f;
}

// ---------------------------------------------------------------------------
// Small utility kernels.
// ---------------------------------------------------------------------------
// Fragment-major weight pack: Bp[(kt*Npad + n)*32 + h*16 + j] = W[kt*32+16h+j][n]
__global__ void pack_w_kernel(const float* __restrict__ W, __bf16* __restrict__ Bp,
                              int Cin, int Cout, int Kpad, int Npad)
{
    int idx = blockIdx.x * blockDim.x + threadIdx.x;
    if (idx >= Kpad * Npad) return;
    int j  = idx & 15;
    int h  = (idx >> 4) & 1;
    int t  = idx >> 5;               // kt*Npad + n
    int n  = t % Npad;
    int kt = t / Npad;
    int kk = kt * 32 + h * 16 + j;
    float v = (kk < Cin && n < Cout) ? W[(size_t)kk * Cout + n] : 0.f;
    Bp[idx] = (__bf16)v;
}

__global__ void dup_xyz_kernel(const float* __restrict__ xyz, float* __restrict__ out,
                               long long npts)
{
    long long i = (long long)blockIdx.x * blockDim.x + threadIdx.x;
    if (i >= npts) return;
    float x = xyz[i * 3], y = xyz[i * 3 + 1], z = xyz[i * 3 + 2];
    float* o = out + i * 6;
    o[0] = x; o[1] = y; o[2] = z; o[3] = x; o[4] = y; o[5] = z;
}

__global__ void f32_to_bf16_kernel(const float* __restrict__ src, __bf16* __restrict__ dst,
                                   long long count)
{
    long long i = (long long)blockIdx.x * blockDim.x + threadIdx.x;
    if (i < count) dst[i] = (__bf16)src[i];
}

__global__ void head_write_kernel(const float* __restrict__ Y, float* __restrict__ out,
                                  long long M, int Npad, int C)
{
    long long idx = (long long)blockIdx.x * blockDim.x + threadIdx.x;
    if (idx >= M * (long long)C) return;
    long long r = idx / C; int c = (int)(idx % C);
    out[(size_t)r * C + c] = Y[(size_t)r * Npad + c];
}

__global__ void head_logsoftmax_kernel(const float* __restrict__ Y, float* __restrict__ out,
                                       long long M, int Npad, int C)
{
    long long r = (long long)blockIdx.x * blockDim.x + threadIdx.x;
    if (r >= M) return;
    const float* yr = Y + (size_t)r * Npad;
    float mx = -FLT_MAX;
    for (int c = 0; c < C; ++c) { float x = yr[c]; if (x > mx) mx = x; }
    float s = 0.f;
    for (int c = 0; c < C; ++c) s += expf(yr[c] - mx);
    float ls = logf(s) + mx;
    for (int c = 0; c < C; ++c) out[(size_t)r * C + c] = yr[c] - ls;
}

__global__ void head_dim_kernel(const float* __restrict__ Y, float* __restrict__ out,
                                long long M, int Npad)
{
    long long r = (long long)blockIdx.x * blockDim.x + threadIdx.x;
    if (r < M) out[r] = Y[(size_t)r * Npad];
}

// ---------------------------------------------------------------------------
// Host-side orchestration.
// ---------------------------------------------------------------------------
struct LayerP { const float *W, *b, *g, *be; int Cin, Cout, Kpad, Npad; __bf16* Wp; };

static inline size_t alignup(size_t x, size_t a) { return (x + a - 1) & ~(a - 1); }
static inline int    cdiv(long long a, long long b) { return (int)((a + b - 1) / b); }
static inline int    a16(int x) { return (x + 15) & ~15; }
static inline int    a32(int x) { return (x + 31) & ~31; }

static void run_gemm(hipStream_t s, const __bf16* A, const LayerP& L, float* Y, long long Mpad)
{
    int Mtiles = (int)(Mpad / 16);
    dim3 g((unsigned)cdiv(Mtiles, 8), (unsigned)(L.Npad / 16));
    wmma_gemm_kernel<<<g, 256, 0, s>>>(A, L.Wp, L.b, Y, L.Kpad, L.Npad, L.Cout, Mtiles);
}

static void run_downsample(hipStream_t s, const float* xyz, const float* fea,
                           int n, int f, int nc, int k, int bs,
                           const LayerP& L0, const LayerP& L1,
                           float* out_xyz, float* out_fea,
                           __bf16* Areg, float* Yreg, int* knnbuf, int* fpsbuf,
                           float* meanb, float* varb)
{
    int ncols32 = a32(n);
    dim3 gknn(bs, (unsigned)cdiv(n, 16));
    knn_kernel<<<gknn, 256, (size_t)16 * ncols32 * sizeof(float), s>>>(xyz, knnbuf, n, ncols32, k);
    fps_kernel<<<bs, 256, 0, s>>>(xyz, fpsbuf, n, nc);

    long long M = (long long)bs * nc * k;
    long long Mpad = (long long)alignup((size_t)M, 16);
    gather_group_kernel<<<cdiv(Mpad, 256), 256, 0, s>>>(xyz, fea, knnbuf, fpsbuf,
                                                        Areg, out_xyz, n, nc, k, f,
                                                        L0.Kpad, Mpad, bs);
    run_gemm(s, Areg, L0, Yreg, Mpad);
    bn_stats_kernel<<<L0.Cout, 256, 0, s>>>(Yreg, meanb, varb, M, L0.Npad);
    bn_act_bf16_kernel<<<cdiv(Mpad, 256), 256, 0, s>>>(Yreg, meanb, varb, L0.g, L0.be,
                                                       Areg, M, Mpad, L0.Npad, L0.Cout, L1.Kpad);
    run_gemm(s, Areg, L1, Yreg, Mpad);
    bn_stats_kernel<<<L1.Cout, 256, 0, s>>>(Yreg, meanb, varb, M, L1.Npad);
    bn_act_f32_kernel<<<cdiv(M * L1.Cout, 256), 256, 0, s>>>(Yreg, meanb, varb, L1.g, L1.be,
                                                             M, L1.Npad, L1.Cout);
    long long bsnc = (long long)bs * nc;
    max_k_kernel<<<cdiv(bsnc * L1.Cout, 256), 256, 0, s>>>(Yreg, out_fea, bsnc, k,
                                                           L1.Npad, L1.Cout);
}

static void run_upsample(hipStream_t s, const float* xyz1, const float* xyz2,
                         const float* fea1, const float* fea2,
                         int n1, int n2, int f1, int f2, int bs,
                         const LayerP& L0, const LayerP& L1, float* out_fea,
                         __bf16* Areg, float* Yreg, float* meanb, float* varb)
{
    long long M = (long long)bs * n1;
    long long Mpad = (long long)alignup((size_t)M, 16);
    three_interp_kernel<<<cdiv(Mpad, 256), 256, 0, s>>>(xyz1, xyz2, fea1, fea2, Areg,
                                                        n1, n2, f1, f2, L0.Kpad, Mpad, bs);
    run_gemm(s, Areg, L0, Yreg, Mpad);
    bn_stats_kernel<<<L0.Cout, 256, 0, s>>>(Yreg, meanb, varb, M, L0.Npad);
    bn_act_bf16_kernel<<<cdiv(Mpad, 256), 256, 0, s>>>(Yreg, meanb, varb, L0.g, L0.be,
                                                       Areg, M, Mpad, L0.Npad, L0.Cout, L1.Kpad);
    run_gemm(s, Areg, L1, out_fea, Mpad);   // final layer: linear only (Npad == Cout)
}

extern "C" void kernel_launch(void* const* d_in, const int* in_sizes, int n_in,
                              void* d_out, int out_size, void* d_ws, size_t ws_size,
                              hipStream_t stream)
{
    (void)in_sizes; (void)n_in; (void)out_size; (void)ws_size;
    const int BS = 8, N0 = 2048;
    const int NC1 = 1843, K1 = 50, NC2 = 1658, K2 = 40, NC3 = 1492, K3 = 30;

    const float* xyz = (const float*)d_in[0];
    const int mlp_ch[11][3] = {
        {6,19,64},{67,92,128},{131,183,256},          // dn1 dn2 dn3
        {384,256,128},{192,128,64},{70,64,32},        // up3 up2 up1
        {32,16,5},{32,16,3},{32,16,1},{32,16,3},{32,16,3} // pmt mad dim nor loc
    };
    LayerP Ls[11][2];
    {
        int p = 1;
        for (int m = 0; m < 11; ++m)
            for (int l = 0; l < 2; ++l) {
                LayerP& L = Ls[m][l];
                L.W  = (const float*)d_in[p++];
                L.b  = (const float*)d_in[p++];
                L.g  = (const float*)d_in[p++];
                L.be = (const float*)d_in[p++];
                L.Cin = mlp_ch[m][l]; L.Cout = mlp_ch[m][l + 1];
                L.Kpad = a32(L.Cin);  L.Npad = a16(L.Cout);
            }
    }

    char* wp = (char*)d_ws;
    auto alloc = [&](size_t bytes) -> char* { char* r = wp; wp += alignup(bytes, 256); return r; };

    float* meanb = (float*)alloc(256 * sizeof(float));
    float* varb  = (float*)alloc(256 * sizeof(float));
    int*  fpsbuf = (int*)alloc((size_t)BS * 2048 * sizeof(int));
    int*  knnbuf = (int*)alloc((size_t)BS * 2048 * 50 * sizeof(int));

    float* l1_xyz = (float*)alloc((size_t)BS * NC1 * 3 * sizeof(float));
    float* l2_xyz = (float*)alloc((size_t)BS * NC2 * 3 * sizeof(float));
    float* l3_xyz = (float*)alloc((size_t)BS * NC3 * 3 * sizeof(float));
    float* l1_fea = (float*)alloc((size_t)BS * NC1 * 64 * sizeof(float));
    float* l2_fea = (float*)alloc((size_t)BS * NC2 * 128 * sizeof(float));
    float* l3_fea = (float*)alloc((size_t)BS * NC3 * 256 * sizeof(float));
    float* l2_fea_new = (float*)alloc(alignup((size_t)BS * NC2, 16) * 128 * sizeof(float));
    float* l1_fea_new = (float*)alloc(alignup((size_t)BS * NC1, 16) * 64 * sizeof(float));
    float* l0_fea     = (float*)alloc((size_t)BS * N0 * 32 * sizeof(float));
    float* xyzdup     = (float*)alloc((size_t)BS * N0 * 6 * sizeof(float));

    for (int m = 0; m < 11; ++m)
        for (int l = 0; l < 2; ++l) {
            LayerP& L = Ls[m][l];
            L.Wp = (__bf16*)alloc((size_t)L.Kpad * L.Npad * sizeof(__bf16));
            pack_w_kernel<<<cdiv((long long)L.Kpad * L.Npad, 256), 256, 0, stream>>>(
                L.W, L.Wp, L.Cin, L.Cout, L.Kpad, L.Npad);
        }

    __bf16* Areg = (__bf16*)alloc((size_t)358080 * 192 * sizeof(__bf16));
    float*  Yreg = (float*)alloc((size_t)358080 * 256 * sizeof(float));

    run_downsample(stream, xyz,    xyz,    N0,  3,   NC1, K1, BS, Ls[0][0], Ls[0][1],
                   l1_xyz, l1_fea, Areg, Yreg, knnbuf, fpsbuf, meanb, varb);
    run_downsample(stream, l1_xyz, l1_fea, NC1, 64,  NC2, K2, BS, Ls[1][0], Ls[1][1],
                   l2_xyz, l2_fea, Areg, Yreg, knnbuf, fpsbuf, meanb, varb);
    run_downsample(stream, l2_xyz, l2_fea, NC2, 128, NC3, K3, BS, Ls[2][0], Ls[2][1],
                   l3_xyz, l3_fea, Areg, Yreg, knnbuf, fpsbuf, meanb, varb);

    run_upsample(stream, l2_xyz, l3_xyz, l2_fea, l3_fea, NC2, NC3, 128, 256, BS,
                 Ls[3][0], Ls[3][1], l2_fea_new, Areg, Yreg, meanb, varb);
    run_upsample(stream, l1_xyz, l2_xyz, l1_fea, l2_fea_new, NC1, NC2, 64, 128, BS,
                 Ls[4][0], Ls[4][1], l1_fea_new, Areg, Yreg, meanb, varb);
    dup_xyz_kernel<<<cdiv((long long)BS * N0, 256), 256, 0, stream>>>(xyz, xyzdup, (long long)BS * N0);
    run_upsample(stream, xyz, l1_xyz, xyzdup, l1_fea_new, N0, NC1, 6, 64, BS,
                 Ls[5][0], Ls[5][1], l0_fea, Areg, Yreg, meanb, varb);

    const long long Mh = (long long)BS * N0;       // 16384 rows
    __bf16* Ah0 = Areg;
    __bf16* Ah1 = Areg + (size_t)Mh * 32;
    f32_to_bf16_kernel<<<cdiv(Mh * 32, 256), 256, 0, stream>>>(l0_fea, Ah0, Mh * 32);

    const int    head_idx[5] = {6, 7, 8, 9, 10};               // pmt mad dim nor loc
    const int    head_c[5]   = {5, 3, 1, 3, 3};
    const size_t head_off[5] = {0, 81920, 131072, 147456, 196608};
    float* out = (float*)d_out;

    for (int h = 0; h < 5; ++h) {
        const LayerP& H0 = Ls[head_idx[h]][0];
        const LayerP& H1 = Ls[head_idx[h]][1];
        run_gemm(stream, Ah0, H0, Yreg, Mh);
        bn_stats_kernel<<<H0.Cout, 256, 0, stream>>>(Yreg, meanb, varb, Mh, H0.Npad);
        bn_act_bf16_kernel<<<cdiv(Mh, 256), 256, 0, stream>>>(Yreg, meanb, varb, H0.g, H0.be,
                                                              Ah1, Mh, Mh, H0.Npad, H0.Cout, H1.Kpad);
        run_gemm(stream, Ah1, H1, Yreg, Mh);
        float* o = out + head_off[h];
        if (h == 0)
            head_logsoftmax_kernel<<<cdiv(Mh, 256), 256, 0, stream>>>(Yreg, o, Mh, H1.Npad, head_c[h]);
        else if (h == 2)
            head_dim_kernel<<<cdiv(Mh, 256), 256, 0, stream>>>(Yreg, o, Mh, H1.Npad);
        else
            head_write_kernel<<<cdiv(Mh * head_c[h], 256), 256, 0, stream>>>(Yreg, o, Mh, H1.Npad, head_c[h]);
    }
}
